// VariableSelection_14396730377069
// MI455X (gfx1250) — matrix operand order repaired and checked by reading, not compiled, not run on hardware
//
#include <hip/hip_runtime.h>
#include <hip/hip_bf16.h>
#include <math.h>

typedef __attribute__((ext_vector_type(16))) _Float16 v16h;
typedef __attribute__((ext_vector_type(8)))  _Float16 v8h;
typedef __attribute__((ext_vector_type(8)))  float    v8f;

#define B_   64
#define T_   512
#define BT_  32768
#define MX_  32
#define IN_  64
#define HID_ 64
#define DD_  2048
#define EPS_ 1e-5f

// ---------------- WMMA helpers (CDNA5 16x16x32 f16 -> f32) ----------------

__device__ __forceinline__ v8f wmma16(v16h a, v16h b, v8f c) {
  // (neg_a, A, neg_b, B, c_mod, C, reuse_a, reuse_b)
  return __builtin_amdgcn_wmma_f32_16x16x32_f16(false, a, false, b, (short)0, c,
                                                false, false);
}

__device__ __forceinline__ v8f splat8(float v) {
  v8f r;
#pragma unroll
  for (int i = 0; i < 8; ++i) r[i] = v;
  return r;
}

__device__ __forceinline__ float elu1(float x) { return x > 0.f ? x : (__expf(x) - 1.f); }
__device__ __forceinline__ float sigm(float x) { return 1.f / (1.f + __expf(-x)); }

// A fragment (16x32, 16-bit): lane L holds row L%16.
// j<8 : K = kbase + 8*(L/16) + j ; j>=8 : K = kbase + 16 + 8*(L/16) + (j-8)
__device__ __forceinline__ v16h a_frag_f32(const float* rowptr, int kbase, int hi) {
  const float* p0 = rowptr + kbase + 8 * hi;
  const float* p1 = p0 + 16;
  v16h a;
#pragma unroll
  for (int j = 0; j < 8; ++j) { a[j] = (_Float16)p0[j]; a[8 + j] = (_Float16)p1[j]; }
  return a;
}

__device__ __forceinline__ v16h a_frag_l(const _Float16* rowptr, int kbase, int hi) {
  v8h lo = *(const v8h*)(rowptr + kbase + 8 * hi);
  v8h h8 = *(const v8h*)(rowptr + kbase + 16 + 8 * hi);
  return __builtin_shufflevector(lo, h8, 0,1,2,3,4,5,6,7,8,9,10,11,12,13,14,15);
}

// B fragment pre-swizzled in workspace: one contiguous v16h (32B) per lane.
__device__ __forceinline__ v16h b_frag(const _Float16* swz, int f, int lane) {
  return *(const v16h*)(swz + ((long)f * 32 + lane) * 16);
}

// Async global -> LDS copy of one 16-byte chunk (ASYNCcnt-tracked).
__device__ __forceinline__ void async_g2l_b128(unsigned lds_off, const float* gsrc) {
  asm volatile("global_load_async_to_lds_b128 %0, %1, off"
               :: "v"(lds_off), "v"(gsrc) : "memory");
}
__device__ __forceinline__ void wait_async0() {
  asm volatile("s_wait_asynccnt 0x0" ::: "memory");
}

// ---------------- Kernel 0: swizzle weights into f16 B-fragment order ------
// B layout (32x16 16-bit): lane L, elem j: K = 32*kc + 8*(L/16) + j (j<8), 32*kc + 16 + 8*(L/16) + (j-8) (j>=8), N = 16*nt + L%16
__global__ __launch_bounds__(256) void k0_swizzle(const float* __restrict__ src,
                                                  _Float16* __restrict__ dst,
                                                  int K, int N, int mats) {
  int nfn = N >> 4;
  long per_mat = (long)(K >> 5) * nfn * 512;
  long total = per_mat * mats;
  long idx = (long)blockIdx.x * 256 + threadIdx.x;
  if (idx >= total) return;
  int  mat = (int)(idx / per_mat);
  long rem = idx % per_mat;
  int j = (int)(rem & 15);
  int L = (int)((rem >> 4) & 31);
  int f = (int)(rem >> 9);
  int kc = f / nfn, nt = f % nfn;
  int k = 32 * kc + (j < 8 ? 8 * (L >> 4) + j : 16 + 8 * (L >> 4) + (j - 8));
  int n = 16 * nt + (L & 15);
  dst[idx] = (_Float16)src[(long)mat * K * N + (long)k * N + n];
}

// ---------------- Kernel 1: weights GRN -> w_out (pre-softmax) -------------
__global__ __launch_bounds__(32) void k1_weights_grn(
    const float* __restrict__ x,
    const _Float16* __restrict__ w1swz, const _Float16* __restrict__ wskipswz,
    const _Float16* __restrict__ w2swz, const _Float16* __restrict__ wgateswz,
    const float* __restrict__ b1, const float* __restrict__ b2,
    const float* __restrict__ bskip, const float* __restrict__ bgate,
    const float* __restrict__ lng, const float* __restrict__ lnb,
    float* __restrict__ w_out) {
  __shared__ __align__(16) _Float16 hstage[16 * 64];
  __shared__ float fstage[16 * 32];

  int lane = threadIdx.x;
  int hi = lane >> 4, lr = lane & 15;
  long row0 = (long)blockIdx.x * 16;
  const float* rowptr = x + (row0 + lr) * DD_;

  v8f acc[4], skp[2];
#pragma unroll
  for (int nt = 0; nt < 4; ++nt) acc[nt] = splat8(b1[16 * nt + lr]);
#pragma unroll
  for (int nt = 0; nt < 2; ++nt) skp[nt] = splat8(bskip[16 * nt + lr]);

  // Software-pipelined K loop: stage kc+1 operands while WMMA-ing kc.
  v16h a_cur = a_frag_f32(rowptr, 0, hi);
  v16h bc[6];
#pragma unroll
  for (int f = 0; f < 4; ++f) bc[f] = b_frag(w1swz, f, lane);
  bc[4] = b_frag(wskipswz, 0, lane);
  bc[5] = b_frag(wskipswz, 1, lane);

  for (int kc = 0; kc < 64; ++kc) {
    int kn = (kc + 1) & 63;   // wraps on last iteration (harmless reload)
    __builtin_prefetch(rowptr + 32 * kn + 64, 0, 1);    // global_prefetch_b8
    v16h a_nxt = a_frag_f32(rowptr, 32 * kn, hi);
    v16h bn[6];
#pragma unroll
    for (int f = 0; f < 4; ++f) bn[f] = b_frag(w1swz, kn * 4 + f, lane);
    bn[4] = b_frag(wskipswz, kn * 2 + 0, lane);
    bn[5] = b_frag(wskipswz, kn * 2 + 1, lane);

#pragma unroll
    for (int nt = 0; nt < 4; ++nt) acc[nt] = wmma16(a_cur, bc[nt], acc[nt]);
    skp[0] = wmma16(a_cur, bc[4], skp[0]);
    skp[1] = wmma16(a_cur, bc[5], skp[1]);

    a_cur = a_nxt;
#pragma unroll
    for (int f = 0; f < 6; ++f) bc[f] = bn[f];
  }

  // ELU(h1) -> LDS f16 [16][64]
#pragma unroll
  for (int nt = 0; nt < 4; ++nt)
#pragma unroll
    for (int r = 0; r < 8; ++r)
      hstage[(r + 8 * hi) * 64 + 16 * nt + lr] = (_Float16)elu1(acc[nt][r]);
  __syncthreads();

  // fc2: [16,64] @ [64,32]
  v8f a2[2];
#pragma unroll
  for (int nt = 0; nt < 2; ++nt) a2[nt] = splat8(b2[16 * nt + lr]);
#pragma unroll
  for (int kc = 0; kc < 2; ++kc) {
    v16h ah = a_frag_l(hstage + lr * 64, 32 * kc, hi);
#pragma unroll
    for (int nt = 0; nt < 2; ++nt) a2[nt] = wmma16(ah, b_frag(w2swz, kc * 2 + nt, lane), a2[nt]);
  }
  __syncthreads();

  // h2 -> LDS f16 [16][32]
#pragma unroll
  for (int nt = 0; nt < 2; ++nt)
#pragma unroll
    for (int r = 0; r < 8; ++r)
      hstage[(r + 8 * hi) * 32 + 16 * nt + lr] = (_Float16)a2[nt][r];
  __syncthreads();

  // gate: [16,32] @ [32,64], then GLU + skip
  v8f a3[4];
#pragma unroll
  for (int nt = 0; nt < 4; ++nt) a3[nt] = splat8(bgate[16 * nt + lr]);
  {
    v16h ah = a_frag_l(hstage + lr * 32, 0, hi);
#pragma unroll
    for (int nt = 0; nt < 4; ++nt) a3[nt] = wmma16(ah, b_frag(wgateswz, nt, lane), a3[nt]);
  }
#pragma unroll
  for (int nt = 0; nt < 2; ++nt)
#pragma unroll
    for (int r = 0; r < 8; ++r) {
      float v = a3[nt][r], g = a3[nt + 2][r];
      fstage[(r + 8 * hi) * 32 + 16 * nt + lr] = skp[nt][r] + v * sigm(g);
    }
  __syncthreads();

  // LayerNorm over 32 per row; lane pair (lr, lr+16) splits the columns
  {
    const float* rp = fstage + lr * 32;
    float m = 0.f;
#pragma unroll
    for (int c = 0; c < 32; ++c) m += rp[(c + lr) & 31];
    m *= (1.f / 32.f);
    float v = 0.f;
#pragma unroll
    for (int c = 0; c < 32; ++c) { float d = rp[(c + lr) & 31] - m; v += d * d; }
    v *= (1.f / 32.f);
    float rinv = rsqrtf(v + EPS_);
#pragma unroll
    for (int c = 0; c < 16; ++c) {
      int col = 16 * hi + c;
      w_out[(row0 + lr) * MX_ + col] = (rp[col] - m) * rinv * lng[col] + lnb[col];
    }
  }
}

// ---------------- Kernel 2: softmax over T per (b, m) ----------------------
__global__ __launch_bounds__(128) void k2_softmax(const float* __restrict__ w_out,
                                                  float* __restrict__ wts) {
  __shared__ float red[128];
  int bm = blockIdx.x;
  int b = bm >> 5, m = bm & 31;
  int tid = threadIdx.x;
  float vals[4];
  float mx = -3.4e38f;
#pragma unroll
  for (int i = 0; i < 4; ++i) {
    int t = tid + 128 * i;
    vals[i] = w_out[((long)b * T_ + t) * MX_ + m];
    mx = fmaxf(mx, vals[i]);
  }
  red[tid] = mx; __syncthreads();
  for (int s = 64; s > 0; s >>= 1) { if (tid < s) red[tid] = fmaxf(red[tid], red[tid + s]); __syncthreads(); }
  mx = red[0]; __syncthreads();
  float sum = 0.f;
#pragma unroll
  for (int i = 0; i < 4; ++i) { vals[i] = __expf(vals[i] - mx); sum += vals[i]; }
  red[tid] = sum; __syncthreads();
  for (int s = 64; s > 0; s >>= 1) { if (tid < s) red[tid] += red[tid + s]; __syncthreads(); }
  float inv = 1.f / red[0];
#pragma unroll
  for (int i = 0; i < 4; ++i) {
    int t = tid + 128 * i;
    wts[((long)b * T_ + t) * MX_ + m] = vals[i] * inv;
  }
}

// ---------------- Kernel 3: per-variable GRNs + weighted sum ---------------
__global__ __launch_bounds__(256) void k3_var_grns(
    const float* __restrict__ x, const float* __restrict__ wts,
    const _Float16* __restrict__ g1swz, const _Float16* __restrict__ g2swz,
    const _Float16* __restrict__ ggswz,
    const float* __restrict__ gb1, const float* __restrict__ gb2,
    const float* __restrict__ gbg,
    const float* __restrict__ glng, const float* __restrict__ glnb,
    float* __restrict__ out) {
  __shared__ float outacc[16 * 64];
  __shared__ __align__(16) _Float16 hstage[8][16 * 64];
  __shared__ __align__(16) float    ystage[8][16 * 64]; // x-tile, then y-tile

  int tid = threadIdx.x;
  int wave = tid >> 5, lane = tid & 31;
  int hi = lane >> 4, lr = lane & 15;
  long row0 = (long)blockIdx.x * 16;

  for (int i = tid; i < 1024; i += 256) outacc[i] = 0.f;
  __syncthreads();

  _Float16* hst = hstage[wave];
  float*    yst = ystage[wave];
  unsigned  yst_lds = (unsigned)(uintptr_t)yst;   // LDS byte offset (addr[31:0])

  for (int mi = 0; mi < 4; ++mi) {
    int m = wave + 8 * mi;

    // Async DMA the wave's 16x64 f32 x-tile into LDS (256 x 16B chunks).
#pragma unroll
    for (int i = 0; i < 8; ++i) {
      int q = i * 32 + lane;            // chunk id
      int row = q >> 4;                 // 16 chunks per 256B row
      int cf = (q & 15) * 4;            // float offset within row
      const float* gsrc = x + (row0 + row) * DD_ + m * IN_ + cf;
      async_g2l_b128(yst_lds + (unsigned)q * 16, gsrc);
    }
    wait_async0();

    v16h a0 = a_frag_f32(yst + lr * 64, 0, hi);
    v16h a1 = a_frag_f32(yst + lr * 64, 32, hi);
    const _Float16* w1 = g1swz + (long)m * 8 * 512;
    const _Float16* w2 = g2swz + (long)m * 8 * 512;
    const _Float16* wg = ggswz + (long)m * 16 * 512;

    // fc1: [16,64]@[64,64]
    v8f acc[4];
#pragma unroll
    for (int nt = 0; nt < 4; ++nt) {
      acc[nt] = splat8(gb1[m * HID_ + 16 * nt + lr]);
      acc[nt] = wmma16(a0, b_frag(w1, 0 * 4 + nt, lane), acc[nt]);
      acc[nt] = wmma16(a1, b_frag(w1, 1 * 4 + nt, lane), acc[nt]);
    }
#pragma unroll
    for (int nt = 0; nt < 4; ++nt)
#pragma unroll
      for (int r = 0; r < 8; ++r)
        hst[(r + 8 * hi) * 64 + 16 * nt + lr] = (_Float16)elu1(acc[nt][r]);
    __syncthreads();

    // fc2: [16,64]@[64,64]
    v8f a2[4];
#pragma unroll
    for (int nt = 0; nt < 4; ++nt) a2[nt] = splat8(gb2[m * HID_ + 16 * nt + lr]);
#pragma unroll
    for (int kc = 0; kc < 2; ++kc) {
      v16h ah = a_frag_l(hst + lr * 64, 32 * kc, hi);
#pragma unroll
      for (int nt = 0; nt < 4; ++nt) a2[nt] = wmma16(ah, b_frag(w2, kc * 4 + nt, lane), a2[nt]);
    }
    __syncthreads();
#pragma unroll
    for (int nt = 0; nt < 4; ++nt)
#pragma unroll
      for (int r = 0; r < 8; ++r)
        hst[(r + 8 * hi) * 64 + 16 * nt + lr] = (_Float16)a2[nt][r];
    __syncthreads();

    // gate: [16,64]@[64,128], GLU with (col, col+64) pairs, + identity skip
    v16h h0 = a_frag_l(hst + lr * 64, 0, hi);
    v16h h1 = a_frag_l(hst + lr * 64, 32, hi);
#pragma unroll
    for (int p = 0; p < 4; ++p) {
      v8f av = splat8(gbg[m * 128 + 16 * p + lr]);
      v8f ag = splat8(gbg[m * 128 + 64 + 16 * p + lr]);
      av = wmma16(h0, b_frag(wg, 0 * 8 + p, lane), av);
      av = wmma16(h1, b_frag(wg, 1 * 8 + p, lane), av);
      ag = wmma16(h0, b_frag(wg, 0 * 8 + p + 4, lane), ag);
      ag = wmma16(h1, b_frag(wg, 1 * 8 + p + 4, lane), ag);
#pragma unroll
      for (int r = 0; r < 8; ++r) {
        int row = r + 8 * hi, col = 16 * p + lr;
        float xv = yst[row * 64 + col];           // identity skip from LDS tile
        yst[row * 64 + col] = xv + av[r] * sigm(ag[r]);  // same-lane RMW, no race
      }
    }
    __syncthreads();

    // LayerNorm over 64 per row, multiply by softmax weight, accumulate
    {
      const float* rp = yst + lr * 64;
      float mn = 0.f;
#pragma unroll
      for (int c = 0; c < 64; ++c) mn += rp[(c + lane) & 63];
      mn *= (1.f / 64.f);
      float vr = 0.f;
#pragma unroll
      for (int c = 0; c < 64; ++c) { float d = rp[(c + lane) & 63] - mn; vr += d * d; }
      vr *= (1.f / 64.f);
      float rinv = rsqrtf(vr + EPS_);
      float wt = wts[(row0 + lr) * MX_ + m];
#pragma unroll
      for (int c = 0; c < 32; ++c) {
        int col = 32 * hi + ((c + lr) & 31);   // skew to spread LDS banks
        float o = (rp[col] - mn) * rinv * glng[m * HID_ + col] + glnb[m * HID_ + col];
        atomicAdd(&outacc[lr * 64 + col], o * wt);  // ds_add_f32
      }
    }
    __syncthreads();
  }

  for (int i = tid; i < 1024; i += 256) {
    int r = i >> 6, c = i & 63;
    out[(row0 + r) * HID_ + c] = outacc[i];
  }
}

// ---------------- Launch ---------------------------------------------------
extern "C" void kernel_launch(void* const* d_in, const int* in_sizes, int n_in,
                              void* d_out, int out_size, void* d_ws, size_t ws_size,
                              hipStream_t stream) {
  const float* x        = (const float*)d_in[0];
  const float* w_fc1_w  = (const float*)d_in[1];
  const float* w_fc1_b  = (const float*)d_in[2];
  const float* w_fc2_w  = (const float*)d_in[3];
  const float* w_fc2_b  = (const float*)d_in[4];
  const float* w_skip_w = (const float*)d_in[5];
  const float* w_skip_b = (const float*)d_in[6];
  const float* w_gate_w = (const float*)d_in[7];
  const float* w_gate_b = (const float*)d_in[8];
  const float* w_ln_g   = (const float*)d_in[9];
  const float* w_ln_b   = (const float*)d_in[10];
  const float* g_fc1_w  = (const float*)d_in[11];
  const float* g_fc1_b  = (const float*)d_in[12];
  const float* g_fc2_w  = (const float*)d_in[13];
  const float* g_fc2_b  = (const float*)d_in[14];
  const float* g_gate_w = (const float*)d_in[15];
  const float* g_gate_b = (const float*)d_in[16];
  const float* g_ln_g   = (const float*)d_in[17];
  const float* g_ln_b   = (const float*)d_in[18];

  float* out = (float*)d_out;                 // [B,T,HID]
  float* wts = out + (long)BT_ * HID_;        // [B,T,1,MX]

  // workspace layout
  float*    wout_ws = (float*)d_ws;                                   // BT*32 f32
  _Float16* hb = (_Float16*)((char*)d_ws + (size_t)BT_ * MX_ * 4);
  _Float16* w1swz    = hb; hb += 64 * 4 * 512;   // 2048x64
  _Float16* wskipswz = hb; hb += 64 * 2 * 512;   // 2048x32
  _Float16* w2swz    = hb; hb += 2 * 2 * 512;    // 64x32
  _Float16* wgateswz = hb; hb += 1 * 4 * 512;    // 32x64
  _Float16* g1swz    = hb; hb += 32L * 8 * 512;  // 32 x 64x64
  _Float16* g2swz    = hb; hb += 32L * 8 * 512;  // 32 x 64x64
  _Float16* ggswz    = hb;                       // 32 x 64x128

  auto swz = [&](const float* s, _Float16* d, int K, int N, int mats) {
    long total = (long)mats * (K / 32) * (N / 16) * 512;
    k0_swizzle<<<(int)((total + 255) / 256), 256, 0, stream>>>(s, d, K, N, mats);
  };
  swz(w_fc1_w,  w1swz,    2048, 64, 1);
  swz(w_skip_w, wskipswz, 2048, 32, 1);
  swz(w_fc2_w,  w2swz,    64,   32, 1);
  swz(w_gate_w, wgateswz, 32,   64, 1);
  swz(g_fc1_w,  g1swz,    64,   64, 32);
  swz(g_fc2_w,  g2swz,    64,   64, 32);
  swz(g_gate_w, ggswz,    64,  128, 32);

  k1_weights_grn<<<BT_ / 16, 32, 0, stream>>>(
      x, w1swz, wskipswz, w2swz, wgateswz,
      w_fc1_b, w_fc2_b, w_skip_b, w_gate_b, w_ln_g, w_ln_b, wout_ws);

  k2_softmax<<<B_ * MX_, 128, 0, stream>>>(wout_ws, wts);

  k3_var_grns<<<BT_ / 16, 256, 0, stream>>>(
      x, wts, g1swz, g2swz, ggswz,
      g_fc1_b, g_fc2_b, g_gate_b, g_ln_g, g_ln_b, out);
}
